// ArgumentClassification_77300821393492
// MI455X (gfx1250) — compile-verified, matrix operand order checked by reading, not run on hardware
//
#include <hip/hip_runtime.h>

typedef __attribute__((ext_vector_type(16))) __bf16 v16bf;
typedef __attribute__((ext_vector_type(8)))  __bf16 v8bf;
typedef __attribute__((ext_vector_type(8)))  float  v8f;

#define BATCH 32
#define SEQ   256
#define EMB   768
#define HID   512
#define GATES 2048   // 4*HID
#define NCLS  30
#define K0P   800    // IN0=770 padded to multiple of 32
#define M_ALL 8192   // SEQ*BATCH

// ---------------------------------------------------------------------------
// helpers
// ---------------------------------------------------------------------------
__device__ __forceinline__ float sigf(float x) { return 1.f / (1.f + __expf(-x)); }

// Load one 16x32 bf16 WMMA operand fragment for this lane from a row-major
// matrix whose row pointer is `row` (row = lane%16 of the tile) and k-offset kk.
// Per ISA: elems 0..7 = K [kk+8*half, +7], elems 8..15 = K [kk+16+8*half, +7].
__device__ __forceinline__ v16bf load_frag(const __bf16* __restrict__ row, int kk, int half) {
  v8bf lo = *reinterpret_cast<const v8bf*>(row + kk + 8 * half);
  v8bf hi = *reinterpret_cast<const v8bf*>(row + kk + 16 + 8 * half);
  v16bf r;
#pragma unroll
  for (int i = 0; i < 8; ++i) { r[i] = lo[i]; r[i + 8] = hi[i]; }
  return r;
}

__device__ __forceinline__ v8f wmma_bf16(v16bf a, v16bf b, v8f c) {
  return __builtin_amdgcn_wmma_f32_16x16x32_bf16(false, a, false, b, (short)0, c, false, false);
}

// Global grid-wide barrier (persistent kernel), monotonically increasing count.
__device__ __forceinline__ void gbar(unsigned* cnt, unsigned k) {
  __syncthreads();
  if (threadIdx.x == 0) {
    __threadfence();
    atomicAdd(cnt, 1u);
    const unsigned target = k * gridDim.x;
    while (__hip_atomic_load(cnt, __ATOMIC_RELAXED, __HIP_MEMORY_SCOPE_AGENT) < target) {
      __builtin_amdgcn_s_sleep(2);
    }
    __threadfence();
  }
  __syncthreads();
}

// ---------------------------------------------------------------------------
// init: zero the two barrier counters (must be stateless across graph replays)
// ---------------------------------------------------------------------------
__global__ void k_init(unsigned* cnt) {
  if (threadIdx.x < 2) cnt[threadIdx.x] = 0u;
}

// ---------------------------------------------------------------------------
// feature prep 1: mean of last-4 hidden states -> x0[:, :768] (bf16), and
// per-token mean (for delta feature). grid = B*S blocks, 256 threads.
// x0 layout: [S][B][K0P] row-major (GEMM A-matrix, m = s*B + b)
// ---------------------------------------------------------------------------
__global__ void k_feat_mean(const float* __restrict__ hs, __bf16* __restrict__ x0,
                            float* __restrict__ meanword) {
  const int bs = blockIdx.x;
  const int b = bs >> 8, s = bs & 255;
  __shared__ float red[256];
  const size_t L = (size_t)BATCH * SEQ * EMB;
  float sum = 0.f;
  for (int e = threadIdx.x; e < EMB; e += 256) {
    size_t off = ((size_t)b * SEQ + s) * EMB + e;
    float v = 0.25f * (hs[off] + hs[off + L] + hs[off + 2 * L] + hs[off + 3 * L]);
    x0[((size_t)s * BATCH + b) * K0P + e] = (__bf16)v;
    sum += v;
  }
  red[threadIdx.x] = sum;
  __syncthreads();
  for (int o = 128; o; o >>= 1) {
    if (threadIdx.x < o) red[threadIdx.x] += red[threadIdx.x + o];
    __syncthreads();
  }
  if (threadIdx.x == 0) meanword[b * SEQ + s] = red[0] * (1.f / (float)EMB);
}

// ---------------------------------------------------------------------------
// feature prep 2: argmax(predicates) (first max), delta, role mask, K padding.
// grid = B blocks, 256 threads (one per s).
// ---------------------------------------------------------------------------
__global__ void k_feat_aux(const int* __restrict__ roles, const int* __restrict__ preds,
                           const float* __restrict__ meanword, __bf16* __restrict__ x0) {
  const int b = blockIdx.x, s = threadIdx.x;
  __shared__ int red[256];
  const int p = preds[b * SEQ + s];
  red[s] = p;
  __syncthreads();
  for (int o = 128; o; o >>= 1) { if (s < o) red[s] = max(red[s], red[s + o]); __syncthreads(); }
  const int mx = red[0];
  __syncthreads();
  red[s] = (p == mx) ? s : 0x7fffffff;
  __syncthreads();
  for (int o = 128; o; o >>= 1) { if (s < o) red[s] = min(red[s], red[s + o]); __syncthreads(); }
  const int idx = red[0];

  const float mp = meanword[b * SEQ + idx];
  const float delta = meanword[b * SEQ + s] - mp;
  const int r = roles[b * SEQ + s];
  const float mask = (r != 0 && r != -100) ? 1.f : 0.f;
  __bf16* row = x0 + ((size_t)s * BATCH + b) * K0P;
  row[768] = (__bf16)delta;
  row[769] = (__bf16)mask;
  for (int k = 770; k < K0P; ++k) row[k] = (__bf16)0.f;
}

// ---------------------------------------------------------------------------
// fp32 -> bf16 pack with zero padding in K and extra zero rows.
// ---------------------------------------------------------------------------
__global__ void k_pack(const float* __restrict__ src, __bf16* __restrict__ dst,
                       int rows_in, int kin, int kout, int rows_out) {
  const size_t total = (size_t)rows_out * kout;
  for (size_t i = (size_t)blockIdx.x * blockDim.x + threadIdx.x; i < total;
       i += (size_t)gridDim.x * blockDim.x) {
    const int r = (int)(i / kout), k = (int)(i % kout);
    const float v = (r < rows_in && k < kin) ? src[(size_t)r * kin + k] : 0.f;
    dst[i] = (__bf16)v;
  }
}

// ---------------------------------------------------------------------------
// WMMA bf16 GEMM: C[M][ldc] = A[M][K] * Bt[N][K]^T  (fp32 accumulate)
// grid.x = M/16, grid.y = N/256, block = 128 (4 waves).
// Each wave computes a 16x64 strip: one A fragment feeds 4 B fragments / 4
// WMMAs per k-step (4x A reuse, 4 independent WMMAs in flight).
// ---------------------------------------------------------------------------
__global__ void k_gemm(const __bf16* __restrict__ A, const __bf16* __restrict__ Bt,
                       float* __restrict__ C, int K, int ldc) {
  const int wave = threadIdx.x >> 5, lane = threadIdx.x & 31;
  const int half = lane >> 4, lm = lane & 15;
  const int m0 = blockIdx.x * 16;
  const int n0 = blockIdx.y * 256 + wave * 64;
  const __bf16* arow = A + (size_t)(m0 + lm) * K;
  const __bf16* brow = Bt + (size_t)(n0 + lm) * K;
  v8f acc[4] = {{}, {}, {}, {}};
  for (int kk = 0; kk < K; kk += 32) {
    v16bf a = load_frag(arow, kk, half);
#pragma unroll
    for (int ni = 0; ni < 4; ++ni) {
      v16bf b = load_frag(brow + (size_t)(ni * 16) * K, kk, half);
      acc[ni] = wmma_bf16(a, b, acc[ni]);
    }
  }
#pragma unroll
  for (int ni = 0; ni < 4; ++ni)
#pragma unroll
    for (int r = 0; r < 8; ++r)
      C[(size_t)(m0 + r + 8 * half) * ldc + n0 + ni * 16 + lm] = acc[ni][r];
}

// ---------------------------------------------------------------------------
// Output GEMM: out[b][s][c] = h1[m=s*B+b][:] . W_out[c][:] + b_out[c], c < 30.
// grid = (512, 2), block = 32 (one wave). K = 1024, N tile = 16 (N padded 32).
// ---------------------------------------------------------------------------
__global__ void k_gemm_out(const __bf16* __restrict__ A, const __bf16* __restrict__ Bt,
                           const float* __restrict__ bout, float* __restrict__ out) {
  const int lane = threadIdx.x & 31, half = lane >> 4, lm = lane & 15;
  const int m0 = blockIdx.x * 16;
  const int n0 = blockIdx.y * 16;
  const __bf16* arow = A + (size_t)(m0 + lm) * 1024;
  const __bf16* brow = Bt + (size_t)(n0 + lm) * 1024;
  v8f acc = {};
  for (int kk = 0; kk < 1024; kk += 32) {
    v16bf a = load_frag(arow, kk, half);
    v16bf b = load_frag(brow, kk, half);
    acc = wmma_bf16(a, b, acc);
  }
  const int c = n0 + lm;
  if (c < NCLS) {
    const float bb = bout[c];
#pragma unroll
    for (int r = 0; r < 8; ++r) {
      const int m = m0 + r + 8 * half;       // m = s*B + b
      const int s = m >> 5, bIdx = m & 31;
      out[((size_t)bIdx * SEQ + s) * NCLS + c] = acc[r] + bb;
    }
  }
}

// ---------------------------------------------------------------------------
// Persistent BiLSTM layer scan. grid = 16 blocks (2 dirs x 8 hidden slices),
// block = 256 threads (8 waves). Per step: gates = Y[t] + bias + h_prev @ U^T
// via WMMA (h_prev staged in LDS; one A fragment feeds 4 Whh B fragments),
// pointwise LSTM cell, write h (bf16, double buffered) + output sequence.
// Grid-wide barrier between steps.
//   Y     : [8192][4096] fp32, cols [dir*2048, +2048) for this direction
//   Ubase : [2 dirs][2048][512] bf16 row-major (g-major) — B operand directly
//   hcur  : [2 parity][2 dirs][32][512] bf16
//   cst   : [2 dirs][32][512] fp32
//   hout  : [256][32][1024] bf16  (t-major; feeds next GEMM / output head)
// ---------------------------------------------------------------------------
__global__ void k_scan(const float* __restrict__ Y, const __bf16* __restrict__ Ubase,
                       const float* __restrict__ bih_f, const float* __restrict__ bhh_f,
                       const float* __restrict__ bih_b, const float* __restrict__ bhh_b,
                       __bf16* __restrict__ hcur, float* __restrict__ cst,
                       __bf16* __restrict__ hout, unsigned* __restrict__ cnt) {
  __shared__ __attribute__((aligned(16))) __bf16 hp[BATCH * HID];  // 32 KB
  __shared__ float gl[BATCH * 256];                                // 32 KB
  const int tid = threadIdx.x;
  const int dir = blockIdx.x >> 3;
  const int j0 = (blockIdx.x & 7) * 64;
  const int wave = tid >> 5, lane = tid & 31;
  const int half = lane >> 4, lm = lane & 15;
  const int gi = wave >> 1;   // gate chunk 0..3 (i,f,g,o)
  const int mi = wave & 1;    // batch-row tile

  const float* bih = dir ? bih_b : bih_f;
  const float* bhh = dir ? bhh_b : bhh_f;
  const __bf16* U = Ubase + (size_t)dir * GATES * HID;
  __bf16* h0buf = hcur + (size_t)dir * BATCH * HID;                      // parity 0
  __bf16* h1buf = hcur + (size_t)(2 + dir) * BATCH * HID;                // parity 1
  float* myc = cst + (size_t)dir * BATCH * HID;

  // zero initial state for this block's hidden slice
  for (int i = tid; i < BATCH * 64; i += 256) {
    const int b = i >> 6, j = j0 + (i & 63);
    h0buf[b * HID + j] = (__bf16)0.f;
    myc[b * HID + j] = 0.f;
  }
  gbar(cnt, 1u);

  // per-wave invariants for the gate GEMM
  const int gbase = gi * HID + j0;                    // gate row base in [0,2048)
  const __bf16* brow = U + (size_t)(gbase + lm) * HID;
  const __bf16* arow = &hp[(mi * 16 + lm) * HID];

  for (int step = 0; step < SEQ; ++step) {
    const int par = step & 1;
    const int t = dir ? (SEQ - 1 - step) : step;
    const __bf16* hsrc = par ? h1buf : h0buf;
    __bf16* hdst = par ? h0buf : h1buf;

    // stage h_prev[32][512] into LDS
    for (int i = tid * 8; i < BATCH * HID; i += 256 * 8)
      *reinterpret_cast<v8bf*>(&hp[i]) = *reinterpret_cast<const v8bf*>(&hsrc[i]);
    __syncthreads();

    // init 4 accumulators (n-subtiles ni=0..3) from xW + biases
    v8f acc[4];
#pragma unroll
    for (int ni = 0; ni < 4; ++ni) {
      const int gcol = gbase + ni * 16 + lm;
      const float bsum = bih[gcol] + bhh[gcol];
#pragma unroll
      for (int r = 0; r < 8; ++r) {
        const int b = mi * 16 + r + 8 * half;
        acc[ni][r] = Y[((size_t)t * BATCH + b) * 4096 + dir * GATES + gcol] + bsum;
      }
    }
    // recurrent GEMM: one LDS A fragment feeds 4 Whh B fragments per k-step
#pragma unroll
    for (int kk = 0; kk < HID; kk += 32) {
      v16bf a = load_frag(arow, kk, half);
#pragma unroll
      for (int ni = 0; ni < 4; ++ni) {
        v16bf b = load_frag(brow + (size_t)(ni * 16) * HID, kk, half);
        acc[ni] = wmma_bf16(a, b, acc[ni]);
      }
    }
#pragma unroll
    for (int ni = 0; ni < 4; ++ni)
#pragma unroll
      for (int r = 0; r < 8; ++r) {
        const int b = mi * 16 + r + 8 * half;
        gl[b * 256 + gi * 64 + ni * 16 + lm] = acc[ni][r];
      }
    __syncthreads();

    // pointwise LSTM cell for this slice (32x64 elements)
    for (int i = tid; i < BATCH * 64; i += 256) {
      const int b = i >> 6, jj = i & 63, j = j0 + jj;
      const float gI = gl[b * 256 + jj];
      const float gF = gl[b * 256 + 64 + jj];
      const float gG = gl[b * 256 + 128 + jj];
      const float gO = gl[b * 256 + 192 + jj];
      float c = myc[b * HID + j];
      c = sigf(gF) * c + sigf(gI) * tanhf(gG);
      const float h = sigf(gO) * tanhf(c);
      myc[b * HID + j] = c;
      hdst[b * HID + j] = (__bf16)h;
      hout[((size_t)t * BATCH + b) * 1024 + dir * HID + j] = (__bf16)h;
    }
    gbar(cnt, (unsigned)(step + 2));
  }
}

// ---------------------------------------------------------------------------
// host side
// ---------------------------------------------------------------------------
extern "C" void kernel_launch(void* const* d_in, const int* in_sizes, int n_in,
                              void* d_out, int out_size, void* d_ws, size_t ws_size,
                              hipStream_t stream) {
  (void)in_sizes; (void)n_in; (void)out_size; (void)ws_size;
  const float* hs    = (const float*)d_in[0];
  const int*   roles = (const int*)d_in[1];
  const int*   preds = (const int*)d_in[2];
  const float* Wih_l0f = (const float*)d_in[3];
  const float* Whh_l0f = (const float*)d_in[4];
  const float* bih_l0f = (const float*)d_in[5];
  const float* bhh_l0f = (const float*)d_in[6];
  const float* Wih_l0b = (const float*)d_in[7];
  const float* Whh_l0b = (const float*)d_in[8];
  const float* bih_l0b = (const float*)d_in[9];
  const float* bhh_l0b = (const float*)d_in[10];
  const float* Wih_l1f = (const float*)d_in[11];
  const float* Whh_l1f = (const float*)d_in[12];
  const float* bih_l1f = (const float*)d_in[13];
  const float* bhh_l1f = (const float*)d_in[14];
  const float* Wih_l1b = (const float*)d_in[15];
  const float* Whh_l1b = (const float*)d_in[16];
  const float* bih_l1b = (const float*)d_in[17];
  const float* bhh_l1b = (const float*)d_in[18];
  const float* W_out = (const float*)d_in[19];
  const float* b_out = (const float*)d_in[20];
  float* out = (float*)d_out;

  // workspace layout (256B aligned slabs)
  char* base = (char*)d_ws;
  size_t off = 0;
  auto take = [&](size_t bytes) -> char* {
    char* p = base + off;
    off += (bytes + 255) & ~(size_t)255;
    return p;
  };
  unsigned* cnt      = (unsigned*)take(256);
  __bf16*   hcur     = (__bf16*)take((size_t)2 * 2 * BATCH * HID * 2);
  float*    cstate   = (float*)take((size_t)2 * BATCH * HID * 4);
  __bf16*   x0       = (__bf16*)take((size_t)M_ALL * K0P * 2);
  __bf16*   Wc0      = (__bf16*)take((size_t)2 * GATES * K0P * 2);
  __bf16*   Wc1      = (__bf16*)take((size_t)2 * GATES * 1024 * 2);
  __bf16*   Upk      = (__bf16*)take((size_t)4 * GATES * HID * 2);
  __bf16*   WoutPk   = (__bf16*)take((size_t)32 * 1024 * 2);
  float*    meanword = (float*)take((size_t)BATCH * SEQ * 4);
  float*    Y        = (float*)take((size_t)M_ALL * 4096 * 4);
  __bf16*   h0       = (__bf16*)take((size_t)M_ALL * 1024 * 2);
  __bf16*   h1       = (__bf16*)take((size_t)M_ALL * 1024 * 2);

  k_init<<<1, 32, 0, stream>>>(cnt);
  k_feat_mean<<<BATCH * SEQ, 256, 0, stream>>>(hs, x0, meanword);
  k_feat_aux<<<BATCH, 256, 0, stream>>>(roles, preds, meanword, x0);

  // weight packing (fp32 -> bf16, pad K, zero extra rows)
  const dim3 pg(512), pb(256);
  k_pack<<<pg, pb, 0, stream>>>(Wih_l0f, Wc0,                        GATES, 770, K0P, GATES);
  k_pack<<<pg, pb, 0, stream>>>(Wih_l0b, Wc0 + (size_t)GATES * K0P,  GATES, 770, K0P, GATES);
  k_pack<<<pg, pb, 0, stream>>>(Wih_l1f, Wc1,                        GATES, 1024, 1024, GATES);
  k_pack<<<pg, pb, 0, stream>>>(Wih_l1b, Wc1 + (size_t)GATES * 1024, GATES, 1024, 1024, GATES);
  k_pack<<<pg, pb, 0, stream>>>(Whh_l0f, Upk + 0 * (size_t)GATES * HID, GATES, HID, HID, GATES);
  k_pack<<<pg, pb, 0, stream>>>(Whh_l0b, Upk + 1 * (size_t)GATES * HID, GATES, HID, HID, GATES);
  k_pack<<<pg, pb, 0, stream>>>(Whh_l1f, Upk + 2 * (size_t)GATES * HID, GATES, HID, HID, GATES);
  k_pack<<<pg, pb, 0, stream>>>(Whh_l1b, Upk + 3 * (size_t)GATES * HID, GATES, HID, HID, GATES);
  k_pack<<<pg, pb, 0, stream>>>(W_out, WoutPk, NCLS, 1024, 1024, 32);

  // layer 0: input projection (both dirs fused, N=4096) then recurrent scan
  k_gemm<<<dim3(M_ALL / 16, 4096 / 256), 128, 0, stream>>>(x0, Wc0, Y, K0P, 4096);
  k_scan<<<16, 256, 0, stream>>>(Y, Upk, bih_l0f, bhh_l0f, bih_l0b, bhh_l0b,
                                 hcur, cstate, h0, cnt + 0);

  // layer 1
  k_gemm<<<dim3(M_ALL / 16, 4096 / 256), 128, 0, stream>>>(h0, Wc1, Y, 1024, 4096);
  k_scan<<<16, 256, 0, stream>>>(Y, Upk + 2 * (size_t)GATES * HID, bih_l1f, bhh_l1f,
                                 bih_l1b, bhh_l1b, hcur, cstate, h1, cnt + 1);

  // output head
  k_gemm_out<<<dim3(M_ALL / 16, 2), 32, 0, stream>>>(h1, WoutPk, b_out, out);
}
